// _MCLSTMCell_79680233275988
// MI455X (gfx1250) — matrix operand order repaired and checked by reading, not compiled, not run on hardware
//
#include <hip/hip_runtime.h>
#include <math.h>

// ---------------- problem constants ----------------
#define T_    2048
#define B_    256
#define H_    64
#define AX_   32
#define GIN_  97
#define KP1   128      // GIN padded to 128 (4 x K32 WMMA steps)
#define BOT_  256
#define NR_   4096     // H*H
#define NWG   16       // persistent workgroups (1 per WGP)
#define NSL   256      // Wr2 output-column slice per WG
#define THREADS 256    // 8 wave32 per WG

typedef _Float16 half_t;
typedef half_t v16h __attribute__((ext_vector_type(16)));
typedef half_t h8   __attribute__((ext_vector_type(8)));
typedef float  v8f  __attribute__((ext_vector_type(8)));

union v16h_u { v16h v; h8 h[2]; };

// ---------------- LDS layout (bytes, all 16B aligned) ----------------
#define LD_F  136   // f tile row stride (f16 elems, mult of 8)
#define LD_H  264   // h tile row stride (f16 elems, mult of 8)
#define LD_R  264   // relu(rpre) f32 row stride

#define O_FT    0                 // [16][LD_F] f16   = 4352
#define O_HT    4352              // [16][LD_H] f16   = 8448
#define O_RR    12800             // [16][LD_R] f32   = 16896
#define O_CQ    29696             // [16][64]  f32    = 4096
#define O_RS4   33792             // [16][4]   f32    = 256
#define O_GI    34048             // [16][64]  f32    = 4096
#define O_GO    38144             // [16][64]  f32    = 4096
#define O_OB    42240             // [16][64]  f32    = 4096
#define O_MIN   46336             // [16][64]  f32    = 4096
#define O_RS1   50432             // [16]      f32    = 64
#define SMEM_BYTES 50496

// ---------------- WMMA fragment loaders (per ISA 7.12.2 layouts) ------------
// A tile (16 x 32 f16): lane l -> row m=l&15, half hh=l>>4.
// Frag elems 0..7  = row m, K = kb + hh*8 .. +7   (contiguous 16B)
// Frag elems 8..15 = row m, K = kb + 16 + hh*8 .. (contiguous 16B)
__device__ __forceinline__ v16h frag_a(const half_t* base, int ld, int kb) {
  const int lane = threadIdx.x & 31;
  const int m = lane & 15, hh = lane >> 4;
  const half_t* p = base + m * ld + kb + hh * 8;
  v16h_u u;
  u.h[0] = *(const h8*)(p);
  u.h[1] = *(const h8*)(p + 16);
  return u.v;
}

// B tile (32 x 16 f16), weights stored transposed [N][K]: same K order as A:
// elems 0..7 K = kb + hh*8 .., elems 8..15 K = kb + 16 + hh*8 ..
__device__ __forceinline__ v16h frag_b(const half_t* baseT, int ld, int n0, int kb) {
  const int lane = threadIdx.x & 31;
  const int n = lane & 15, hh = lane >> 4;
  const half_t* p = baseT + (size_t)(n0 + n) * ld + kb + hh * 8;
  v16h_u u;
  u.h[0] = *(const h8*)(p);
  u.h[1] = *(const h8*)(p + 16);
  return u.v;
}

// ---------------- deterministic grid barrier (agent scope) ----------------
__device__ __forceinline__ void gbarrier(unsigned* cnt, unsigned* gen, unsigned& lg) {
  __syncthreads();
  if (threadIdx.x == 0) {
    const unsigned g = ++lg;
    __builtin_amdgcn_fence(__ATOMIC_RELEASE, "agent");
    const unsigned prev =
        __hip_atomic_fetch_add(cnt, 1u, __ATOMIC_ACQ_REL, __HIP_MEMORY_SCOPE_AGENT);
    if (prev == NWG - 1) {
      __hip_atomic_store(cnt, 0u, __ATOMIC_RELAXED, __HIP_MEMORY_SCOPE_AGENT);
      __hip_atomic_store(gen, g, __ATOMIC_RELEASE, __HIP_MEMORY_SCOPE_AGENT);
    } else {
      while (__hip_atomic_load(gen, __ATOMIC_ACQUIRE, __HIP_MEMORY_SCOPE_AGENT) < g)
        __builtin_amdgcn_s_sleep(8);
    }
    __builtin_amdgcn_fence(__ATOMIC_ACQUIRE, "agent");
  }
  __syncthreads();
}

// ---------------- preprocessing kernels ----------------
// out[n][kp] = (kp < K) ? W[kp][n] : 0   (f32 -> f16, transpose + K padding)
__global__ void convT_kernel(const float* __restrict__ W, half_t* __restrict__ out,
                             int K, int N, int KP) {
  int idx = blockIdx.x * blockDim.x + threadIdx.x;
  if (idx >= N * KP) return;
  int n = idx / KP, kp = idx - n * KP;
  float v = (kp < K) ? W[(size_t)kp * N + n] : 0.f;
  out[idx] = (half_t)v;
}

__global__ void zero_kernel(unsigned* __restrict__ p, int n) {
  int idx = blockIdx.x * blockDim.x + threadIdx.x;
  if (idx < n) p[idx] = 0u;
}

// ---------------- persistent recurrent kernel ----------------
__global__ void __launch_bounds__(THREADS)
mclstm_kernel(const float* __restrict__ x_m, const float* __restrict__ x_a,
              const float* __restrict__ bo, const float* __restrict__ bi1,
              const float* __restrict__ bi2, const float* __restrict__ br1,
              const float* __restrict__ br2,
              const half_t* __restrict__ wWot, const half_t* __restrict__ wWi1t,
              const half_t* __restrict__ wWi2t, const half_t* __restrict__ wWr1t,
              const half_t* __restrict__ wWr2t,
              float* cbuf, float* partials, unsigned* bar,
              float* __restrict__ out_m, float* __restrict__ out_c) {
  extern __shared__ char smem[];
  half_t* f_t   = (half_t*)(smem + O_FT);    // [16][LD_F]
  half_t* h_t   = (half_t*)(smem + O_HT);    // [16][LD_H]
  float*  rrel  = (float*)(smem + O_RR);     // [16][LD_R]
  float*  cq    = (float*)(smem + O_CQ);     // [16][64]
  float*  rs4   = (float*)(smem + O_RS4);    // [16][4]
  float*  gi    = (float*)(smem + O_GI);     // [16][64]
  float*  go    = (float*)(smem + O_GO);     // [16][64]
  float*  o_buf = (float*)(smem + O_OB);     // [16][64]
  float*  m_in  = (float*)(smem + O_MIN);    // [16][64]
  float*  rs1   = (float*)(smem + O_RS1);    // [16]

  const int tid  = threadIdx.x;
  const int wg   = blockIdx.x;          // 0..15
  const int lane = tid & 31;
  const int wid  = tid >> 5;            // wave id 0..7

  // ---- B fragments for Wr1 / Wr2-slice: VGPR-resident for the whole scan ----
  // Per wave: 2 n-tiles x (4 + 8) K-frags = 24 x v16h = 192 VGPRs pinned.
  v16h bW1[2][4];
  v16h bW2[2][8];
#pragma unroll
  for (int j = 0; j < 2; ++j) {
    const int n0 = wid * 32 + j * 16;
#pragma unroll
    for (int kk = 0; kk < 4; ++kk)
      bW1[j][kk] = frag_b(wWr1t, KP1, n0, kk * 32);
#pragma unroll
    for (int kk = 0; kk < 8; ++kk)
      bW2[j][kk] = frag_b(wWr2t + (size_t)wg * NSL * BOT_, BOT_, n0, kk * 32);
  }

  unsigned lg = 0;
  unsigned* g_cnt = bar;
  unsigned* g_gen = bar + 1;

  for (int t = 0; t < T_; ++t) {
    if (t + 1 < T_ && tid < 64) {  // prefetch next step's aux inputs into L2
      __builtin_prefetch((const char*)&x_a[((size_t)(t + 1) * B_) * AX_] + tid * 512, 0, 1);
    }
    // =========== phase A: all batch tiles, partial m_sys for our 4 h-rows ===========
    for (int bt = 0; bt < NWG; ++bt) {
      const int b0 = bt * 16;
      {  // load c rows for this batch tile
        const int r = tid >> 4, c4 = (tid & 15) * 4;
        float4 cv = *(const float4*)&cbuf[(size_t)(b0 + r) * H_ + c4];
        *(float4*)&cq[r * 64 + c4] = cv;
      }
      __syncthreads();
      if (tid < 16) {  // cn denominator: sum|c| + 1e-5
        float s = 0.f;
        for (int k = 0; k < 64; ++k) s += fabsf(cq[tid * 64 + k]);
        rs1[tid] = s + 1e-5f;
      }
      __syncthreads();
      {  // build f tile [16][128] f16: [x_m | x_a | cn | 0-pad]
        const int r = tid >> 4, c0 = tid & 15;
        const int bg = b0 + r;
        const float inv = 1.0f / rs1[r];
        for (int col = c0; col < KP1; col += 16) {
          float v;
          if (col == 0)            v = x_m[(size_t)t * B_ + bg];
          else if (col <= AX_)     v = x_a[((size_t)t * B_ + bg) * AX_ + (col - 1)];
          else if (col < GIN_)     v = cq[r * 64 + (col - 33)] * inv;
          else                     v = 0.f;
          f_t[r * LD_F + col] = (half_t)v;
        }
      }
      __syncthreads();

      // ---- GEMM1: h_r = relu(f @ Wr1 + br1), N=256 split 2 tiles/wave ----
      {
        v16h afr[4];
#pragma unroll
        for (int kk = 0; kk < 4; ++kk) afr[kk] = frag_a(f_t, LD_F, kk * 32);
#pragma unroll
        for (int j = 0; j < 2; ++j) {
          const int n0 = wid * 32 + j * 16;
          v8f acc = {};
#pragma unroll
          for (int kk = 0; kk < 4; ++kk)
            acc = __builtin_amdgcn_wmma_f32_16x16x32_f16(false, afr[kk], false, bW1[j][kk],
                                                         (short)0, acc, false, false);
          const int n = lane & 15, mb = (lane >> 4) * 8;
          const float bias = br1[n0 + n];
#pragma unroll
          for (int r = 0; r < 8; ++r) {
            float v = acc[r] + bias;
            h_t[(mb + r) * LD_H + n0 + n] = (half_t)(v > 0.f ? v : 0.f);
          }
        }
      }
      __syncthreads();

      // ---- GEMM2: rpre = h_r @ Wr2slice + br2, relu -> rrel (f32) ----
      {
        v16h afr[8];
#pragma unroll
        for (int kk = 0; kk < 8; ++kk) afr[kk] = frag_a(h_t, LD_H, kk * 32);
#pragma unroll
        for (int j = 0; j < 2; ++j) {
          const int n0 = wid * 32 + j * 16;
          v8f acc = {};
#pragma unroll
          for (int kk = 0; kk < 8; ++kk)
            acc = __builtin_amdgcn_wmma_f32_16x16x32_f16(false, afr[kk], false, bW2[j][kk],
                                                         (short)0, acc, false, false);
          const int n = lane & 15, mb = (lane >> 4) * 8;
          const float bias = br2[wg * NSL + n0 + n];
#pragma unroll
          for (int r = 0; r < 8; ++r) {
            float v = acc[r] + bias;
            rrel[(mb + r) * LD_R + n0 + n] = (v > 0.f ? v : 0.f);
          }
        }
      }
      __syncthreads();

      // L1 row sums over each 64-wide k chunk (4 h-rows per WG slice)
      if (tid < 64) {
        const int r = tid >> 2, j = tid & 3;
        float s = 0.f;
        for (int k = 0; k < 64; ++k) s += rrel[r * LD_R + j * 64 + k];
        rs4[r * 4 + j] = fmaxf(s, 1e-12f);
      }
      __syncthreads();

      // partial m_sys[b,k] = sum_j c[b, 4*wg+j] * r_norm -> global partials[wg]
      {
        const int r = tid >> 4, kq = tid & 15;
        float outv[4] = {0.f, 0.f, 0.f, 0.f};
#pragma unroll
        for (int j = 0; j < 4; ++j) {
          const float cj = cq[r * 64 + wg * 4 + j];
          const float sc = cj / rs4[r * 4 + j];
#pragma unroll
          for (int q = 0; q < 4; ++q)
            outv[q] += sc * rrel[r * LD_R + j * 64 + kq * 4 + q];
        }
        *(float4*)&partials[((size_t)wg * B_ + b0 + r) * H_ + kq * 4] =
            make_float4(outv[0], outv[1], outv[2], outv[3]);
      }

      // ---- own-batch-slice gate MLPs (i, o) ----
      if (bt == wg) {
        __syncthreads();
        {  // h_i = relu(f @ Wi1 + bi1) -> overwrite h_t
          v16h afr[4];
#pragma unroll
          for (int kk = 0; kk < 4; ++kk) afr[kk] = frag_a(f_t, LD_F, kk * 32);
#pragma unroll
          for (int j = 0; j < 2; ++j) {
            const int n0 = wid * 32 + j * 16;
            v8f acc = {};
#pragma unroll
            for (int kk = 0; kk < 4; ++kk) {
              v16h bfr = frag_b(wWi1t, KP1, n0, kk * 32);
              acc = __builtin_amdgcn_wmma_f32_16x16x32_f16(false, afr[kk], false, bfr,
                                                           (short)0, acc, false, false);
            }
            const int n = lane & 15, mb = (lane >> 4) * 8;
            const float bias = bi1[n0 + n];
#pragma unroll
            for (int r = 0; r < 8; ++r) {
              float v = acc[r] + bias;
              h_t[(mb + r) * LD_H + n0 + n] = (half_t)(v > 0.f ? v : 0.f);
            }
          }
        }
        __syncthreads();
        if (wid < 4) {  // ipre = h_i @ Wi2 + bi2 (N=64)
          const int n0 = wid * 16;
          v8f acc = {};
#pragma unroll
          for (int kk = 0; kk < 8; ++kk) {
            v16h afr = frag_a(h_t, LD_H, kk * 32);
            v16h bfr = frag_b(wWi2t, BOT_, n0, kk * 32);
            acc = __builtin_amdgcn_wmma_f32_16x16x32_f16(false, afr, false, bfr,
                                                         (short)0, acc, false, false);
          }
          const int n = lane & 15, mb = (lane >> 4) * 8;
          const float bias = bi2[n0 + n];
#pragma unroll
          for (int r = 0; r < 8; ++r) gi[(mb + r) * 64 + n0 + n] = acc[r] + bias;
        } else {        // opre = f @ Wo + bo (N=64)
          const int n0 = (wid - 4) * 16;
          v8f acc = {};
#pragma unroll
          for (int kk = 0; kk < 4; ++kk) {
            v16h afr = frag_a(f_t, LD_F, kk * 32);
            v16h bfr = frag_b(wWot, KP1, n0, kk * 32);
            acc = __builtin_amdgcn_wmma_f32_16x16x32_f16(false, afr, false, bfr,
                                                         (short)0, acc, false, false);
          }
          const int n = lane & 15, mb = (lane >> 4) * 8;
          const float bias = bo[n0 + n];
#pragma unroll
          for (int r = 0; r < 8; ++r) go[(mb + r) * 64 + n0 + n] = acc[r] + bias;
        }
        __syncthreads();
        {  // sigmoids
          const int r = tid >> 4, kq = tid & 15;
#pragma unroll
          for (int q = 0; q < 4; ++q) {
            const int k = kq * 4 + q;
            gi[r * 64 + k] = 1.f / (1.f + expf(-gi[r * 64 + k]));
            o_buf[r * 64 + k] = 1.f / (1.f + expf(-go[r * 64 + k]));
          }
        }
        __syncthreads();
        if (tid < 16) {  // L1 norm of i over H
          float s = 0.f;
          for (int k = 0; k < 64; ++k) s += gi[tid * 64 + k];
          rs1[tid] = fmaxf(s, 1e-12f);
        }
        __syncthreads();
        {  // m_in = x_m * i_norm
          const int r = tid >> 4, kq = tid & 15;
          const float xm = x_m[(size_t)t * B_ + wg * 16 + r];
          const float sc = xm / rs1[r];
#pragma unroll
          for (int q = 0; q < 4; ++q) {
            const int k = kq * 4 + q;
            m_in[r * 64 + k] = gi[r * 64 + k] * sc;
          }
        }
      }
      __syncthreads();
    }  // bt

    // =========== barrier 1: all partials visible ===========
    gbarrier(g_cnt, g_gen, lg);

    // =========== phase B: finalize own 16-batch slice (deterministic order) ==========
    {
      const int r = tid >> 4, kq = tid & 15;
      const int bg = wg * 16 + r;
      float4 m4 = *(float4*)&m_in[r * 64 + kq * 4];
      float mn[4] = {m4.x, m4.y, m4.z, m4.w};
#pragma unroll
      for (int g = 0; g < NWG; ++g) {
        float4 p4 = *(const float4*)&partials[((size_t)g * B_ + bg) * H_ + kq * 4];
        mn[0] += p4.x; mn[1] += p4.y; mn[2] += p4.z; mn[3] += p4.w;
      }
      float4 ov = *(float4*)&o_buf[r * 64 + kq * 4];
      const float oo[4] = {ov.x, ov.y, ov.z, ov.w};
      float mo[4], ct[4];
#pragma unroll
      for (int q = 0; q < 4; ++q) {
        mo[q] = oo[q] * mn[q];
        ct[q] = (1.f - oo[q]) * mn[q];
      }
      const size_t base = ((size_t)t * B_ + bg) * H_ + kq * 4;
      *(float4*)&out_m[base] = make_float4(mo[0], mo[1], mo[2], mo[3]);
      *(float4*)&out_c[base] = make_float4(ct[0], ct[1], ct[2], ct[3]);
      *(float4*)&cbuf[(size_t)bg * H_ + kq * 4] = make_float4(ct[0], ct[1], ct[2], ct[3]);
    }

    // =========== barrier 2: c update visible before next step ===========
    gbarrier(g_cnt, g_gen, lg);
  }  // t
}

// ---------------- host launch ----------------
extern "C" void kernel_launch(void* const* d_in, const int* in_sizes, int n_in,
                              void* d_out, int out_size, void* d_ws, size_t ws_size,
                              hipStream_t stream) {
  (void)in_sizes; (void)n_in; (void)out_size; (void)ws_size;
  const float* x_m = (const float*)d_in[0];
  const float* x_a = (const float*)d_in[1];
  const float* Wo  = (const float*)d_in[2];
  const float* bo  = (const float*)d_in[3];
  const float* Wi1 = (const float*)d_in[4];
  const float* bi1 = (const float*)d_in[5];
  const float* Wi2 = (const float*)d_in[6];
  const float* bi2 = (const float*)d_in[7];
  const float* Wr1 = (const float*)d_in[8];
  const float* br1 = (const float*)d_in[9];
  const float* Wr2 = (const float*)d_in[10];
  const float* br2 = (const float*)d_in[11];

  char* ws = (char*)d_ws;
  size_t off = 0;
  auto take = [&](size_t bytes) -> char* {
    char* p = ws + off;
    off = (off + bytes + 255) & ~(size_t)255;
    return p;
  };
  half_t* wWr2t = (half_t*)take((size_t)NR_ * BOT_ * sizeof(half_t));
  half_t* wWi1t = (half_t*)take((size_t)BOT_ * KP1 * sizeof(half_t));
  half_t* wWr1t = (half_t*)take((size_t)BOT_ * KP1 * sizeof(half_t));
  half_t* wWi2t = (half_t*)take((size_t)H_ * BOT_ * sizeof(half_t));
  half_t* wWot  = (half_t*)take((size_t)H_ * KP1 * sizeof(half_t));
  float*  cbuf  = (float*)take((size_t)B_ * H_ * sizeof(float));
  float*  parts = (float*)take((size_t)NWG * B_ * H_ * sizeof(float));
  unsigned* bar = (unsigned*)take(2 * sizeof(unsigned));

  const dim3 blk(256);
  // weight transposes f32 -> f16 ([N][K] layout, K zero-padded)
  convT_kernel<<<dim3((NR_ * BOT_ + 255) / 256), blk, 0, stream>>>(Wr2, wWr2t, BOT_, NR_, BOT_);
  convT_kernel<<<dim3((BOT_ * KP1 + 255) / 256), blk, 0, stream>>>(Wi1, wWi1t, GIN_, BOT_, KP1);
  convT_kernel<<<dim3((BOT_ * KP1 + 255) / 256), blk, 0, stream>>>(Wr1, wWr1t, GIN_, BOT_, KP1);
  convT_kernel<<<dim3((H_ * BOT_ + 255) / 256), blk, 0, stream>>>(Wi2, wWi2t, BOT_, H_, BOT_);
  convT_kernel<<<dim3((H_ * KP1 + 255) / 256), blk, 0, stream>>>(Wo, wWot, GIN_, H_, KP1);
  // c0 = 0, barrier state = 0
  zero_kernel<<<dim3((B_ * H_ + 255) / 256), blk, 0, stream>>>((unsigned*)cbuf, B_ * H_);
  zero_kernel<<<dim3(1), blk, 0, stream>>>(bar, 2);

  float* out_m = (float*)d_out;
  float* out_c = out_m + (size_t)T_ * B_ * H_;

  mclstm_kernel<<<dim3(NWG), dim3(THREADS), SMEM_BYTES, stream>>>(
      x_m, x_a, bo, bi1, bi2, br1, br2,
      wWot, wWi1t, wWi2t, wWr1t, wWr2t,
      cbuf, parts, bar, out_m, out_c);
}
